// HMM_8186207666612
// MI455X (gfx1250) — matrix operand order repaired and checked by reading, not compiled, not run on hardware
//
#include <hip/hip_runtime.h>

#define B_  64
#define T_  512
#define S_  4
#define H_  512
#define V_  10000

typedef __bf16 bf16;
typedef __attribute__((ext_vector_type(16))) bf16  v16bf;
typedef __attribute__((ext_vector_type(8)))  bf16  v8bf;
typedef __attribute__((ext_vector_type(8)))  float v8f;
typedef unsigned int       u32;
typedef unsigned long long u64;
typedef __attribute__((ext_vector_type(4))) u32 u32x4;
typedef __attribute__((ext_vector_type(8))) u32 u32x8;

// ---------------------------------------------------------------------------
// TDM: async 2D tile load global -> LDS (CDNA5 Tensor Data Mover).
// D# layout per ISA 08_async_tensor §8: group0 = {count, lds_addr,
// global_addr, type=2}; group1 = {data_size, tensor dims, tile dims, stride}.
// 2D tensor -> VADDR2/VADDR3 omitted (NULL).  Tracked by TENSORcnt.
// ---------------------------------------------------------------------------
__device__ __forceinline__ void tdm_load_2d_f32(u32 lds_addr, u64 gaddr,
                                                u32 width, u32 rows,
                                                u32 row_stride) {
  u32x4 g0;
  g0[0] = 1u;                                          // count=1, user mode
  g0[1] = lds_addr;                                    // LDS byte address
  g0[2] = (u32)(gaddr & 0xFFFFFFFFu);                  // global addr [31:0]
  g0[3] = (u32)((gaddr >> 32) & 0x1FFFFFFu)            // global addr [56:32]
        | (2u << 30);                                  // type = 2 ("image")
  u32x8 g1;
  g1[0] = (2u << 16);                                  // data_size=2 -> 4 bytes
  g1[1] = (width & 0xFFFFu) << 16;                     // tensor_dim0 [15:0]
  g1[2] = ((width >> 16) & 0xFFFFu)                    // tensor_dim0 [31:16]
        | ((rows & 0xFFFFu) << 16);                    // tensor_dim1 [15:0]
  g1[3] = ((rows >> 16) & 0xFFFFu)                     // tensor_dim1 [31:16]
        | ((width & 0xFFFFu) << 16);                   // tile_dim0
  g1[4] = (rows & 0xFFFFu);                            // tile_dim1 (tile_dim2=0)
  g1[5] = row_stride;                                  // tensor_dim0_stride lo32
  g1[6] = 0u;                                          // stride hi / dim1_stride
  g1[7] = 0u;
  asm volatile("tensor_load_to_lds %0, %1" :: "s"(g0), "s"(g1) : "memory");
}

// ---------------------------------------------------------------------------
// K1: per-(s,h) logsumexp over vocab V.  2048 rows x 10000 floats; the whole
// 82MB table is L2-resident on MI455X (192MB L2).
// ---------------------------------------------------------------------------
__global__ void emis_lse_kernel(const float* __restrict__ emis,
                                float* __restrict__ lse) {
  const int row = blockIdx.x;                 // s*H + h
  const float* p = emis + (size_t)row * V_;
  __shared__ float red[256];
  const int tx = threadIdx.x;

  float m = -3.4e38f;
  for (int v = tx; v < V_; v += 256) m = fmaxf(m, p[v]);
  red[tx] = m; __syncthreads();
  for (int s = 128; s > 0; s >>= 1) {
    if (tx < s) red[tx] = fmaxf(red[tx], red[tx + s]);
    __syncthreads();
  }
  m = red[0]; __syncthreads();

  float acc = 0.f;
  for (int v = tx; v < V_; v += 256) acc += __expf(p[v] - m);
  red[tx] = acc; __syncthreads();
  for (int s = 128; s > 0; s >>= 1) {
    if (tx < s) red[tx] += red[tx + s];
    __syncthreads();
  }
  if (tx == 0) lse[row] = m + __logf(red[0]);
}

// ---------------------------------------------------------------------------
// K2: P = softmax(tran) rows, stored transposed in bf16 (PT[k][j] = P[j][k])
// so each WMMA B-tile is a 32B-contiguous per-lane load.
// ---------------------------------------------------------------------------
__global__ void tran_softmax_kernel(const float* __restrict__ tr,
                                    bf16* __restrict__ PT) {
  __shared__ float red[256];
  const int j  = blockIdx.x;
  const int tx = threadIdx.x;
  const float* rowp = tr + (size_t)j * H_;

  float x0 = rowp[tx], x1 = rowp[tx + 256];
  float m = fmaxf(x0, x1);
  red[tx] = m; __syncthreads();
  for (int s = 128; s > 0; s >>= 1) {
    if (tx < s) red[tx] = fmaxf(red[tx], red[tx + s]);
    __syncthreads();
  }
  m = red[0]; __syncthreads();

  float e0 = __expf(x0 - m), e1 = __expf(x1 - m);
  red[tx] = e0 + e1; __syncthreads();
  for (int s = 128; s > 0; s >>= 1) {
    if (tx < s) red[tx] += red[tx + s];
    __syncthreads();
  }
  const float inv = 1.f / red[0];
  PT[(size_t)tx          * H_ + j] = (bf16)(e0 * inv);
  PT[(size_t)(tx + 256)  * H_ + j] = (bf16)(e1 * inv);
}

// ---------------------------------------------------------------------------
// K3: eem[t][b][h] = exp( mean_s emis[s,h,obs[b,t,s]] - c[h] ),
// c[h] = mean_s lse[s,h].  (T,B,H) layout -> per-step tiles are contiguous
// 16x512 f32 blocks, ideal for TDM staging in the scan.
// ---------------------------------------------------------------------------
__global__ void emission_kernel(const int* __restrict__ obs,
                                const float* __restrict__ emis,
                                const float* __restrict__ lse,
                                float* __restrict__ eem) {
  const int bt = blockIdx.x;                  // b*T + t
  const int b = bt / T_, t = bt % T_;
  const int v0 = obs[bt * S_ + 0];
  const int v1 = obs[bt * S_ + 1];
  const int v2 = obs[bt * S_ + 2];
  const int v3 = obs[bt * S_ + 3];
  const float* e0 = emis + 0 * (size_t)H_ * V_ + v0;
  const float* e1 = emis + 1 * (size_t)H_ * V_ + v1;
  const float* e2 = emis + 2 * (size_t)H_ * V_ + v2;
  const float* e3 = emis + 3 * (size_t)H_ * V_ + v3;

  for (int h = threadIdx.x; h < H_; h += 256) {
    float s = e0[(size_t)h * V_] + e1[(size_t)h * V_] +
              e2[(size_t)h * V_] + e3[(size_t)h * V_];
    float c = lse[h] + lse[H_ + h] + lse[2 * H_ + h] + lse[3 * H_ + h];
    eem[((size_t)t * B_ + b) * H_ + h] = __expf(0.25f * (s - c));
  }
}

// ---------------------------------------------------------------------------
// K4: sequential forward scan.  4 WGs x 16 batch rows, no cross-WG sync.
// Per step: lin(16x512) = u(16x512,bf16) @ P(512x512,bf16) via
// v_wmma_f32_16x16x32_bf16; then t = lin*eem, r = rowsum, u' = t/r,
// M += log r;  sums[t,b] = M  (== logsumexp of log-alpha, exactly).
//
//  - B residency: each wave keeps n-tile 0's 16 B-tiles (128 VGPRs)
//    register-resident for all steps; n-tile 1 streams from the L2-resident
//    PT each step.  The stream offset passes through an empty asm each
//    iteration so LICM can neither hoist (-> spill) nor upgrade scope:
//    loads stay plain RT-scope b128, clause-able.
//  - eem tiles (16x512 f32 = 32KB/step) are double-buffered into LDS by the
//    Tensor Data Mover (tensor_load_to_lds, TENSORcnt), issued by wave 0 one
//    step ahead, so emission reads never sit on the critical path.
// ---------------------------------------------------------------------------
__global__ __launch_bounds__(512, 1)
void scan_kernel(const bf16* __restrict__ PT,
                 const float* __restrict__ eem,
                 const float* __restrict__ prior,
                 float* __restrict__ sums) {
  __shared__ bf16  U[2][16][H_];        // ping-pong normalized state, 32 KB
  __shared__ float eemLds[2][16][H_];   // TDM-staged emission tiles, 64 KB
  __shared__ float r[16];               // per-row sums
  __shared__ float Mrow[16];            // running log-scale

  const int tid   = threadIdx.x;
  const int lane  = tid & 31;
  const int wid   = tid >> 5;        // 16 waves
  const int half  = (lane >> 4) & 1; // which 16-lane half
  const int ln    = lane & 15;
  const int bbase = blockIdx.x * 16;

  const int col0 = wid * 32 + ln;    // owned output columns (2 N-tiles)
  const int col1 = col0 + 16;
  const float pe0 = __expf(prior[col0]);
  const float pe1 = __expf(prior[col1]);

  const u32 eLds0 = (u32)(size_t)&eemLds[0][0][0];   // LDS byte offsets
  const u32 eLds1 = (u32)(size_t)&eemLds[1][0][0];
  const u64 eGlb  = (u64)(size_t)eem;

  // Resident B half: n-tile 0, all 16 K-blocks (16 x v16bf = 128 VGPRs).
  // B layout (32x16 bf16): lane = N, 16 consecutive K per half-wave.
  v16bf Bres[16];
  #pragma unroll
  for (int kb = 0; kb < 16; ++kb)
    Bres[kb] = *(const v16bf*)(&PT[(size_t)col0 * H_ + kb * 32 + half * 16]);

  // Kick off the t=0 emission tile on the TDM.
  if (wid == 0)
    tdm_load_2d_f32(eLds0, eGlb + ((u64)(0 * B_ + bbase) * H_) * 4, H_, 16, H_);

  if (tid < 16) { r[tid] = 0.f; Mrow[tid] = 0.f; }
  __syncthreads();

  int cur = 0;
  for (int t = 0; t < T_; ++t) {
    // Wave 0: issue next tile, then wait until tile t has landed in LDS.
    if (wid == 0) {
      if (t + 1 < T_) {
        tdm_load_2d_f32((t & 1) ? eLds0 : eLds1,
                        eGlb + ((u64)((t + 1) * B_ + bbase) * H_) * 4,
                        H_, 16, H_);
        __builtin_amdgcn_s_wait_tensorcnt(1);   // tile t done (in-order)
      } else {
        __builtin_amdgcn_s_wait_tensorcnt(0);
      }
    }
    __syncthreads();                            // publish tile t
    const float (*et)[H_] = eemLds[t & 1];

    float tv0[8], tv1[8];

    if (t == 0) {
      // alpha0 = em0 + prior  ->  t = exp(prior) * eem0   (M_prev = 0)
      #pragma unroll
      for (int i = 0; i < 8; ++i) {
        const int m = i + 8 * half;
        tv0[i] = pe0 * et[m][col0];
        tv1[i] = pe1 * et[m][col1];
      }
    } else {
      // Streamed-B base offset, made opaque per iteration: defeats LICM
      // (no hoist -> no spill) while keeping plain RT-scope global loads.
      size_t o1 = (size_t)col1 * H_;
      asm volatile("" : "+v"(o1));
      v8f acc0 = {};
      v8f acc1 = {};
      #pragma unroll
      for (int kb = 0; kb < 16; ++kb) {
        // A-tile (16x32 bf16): lanes 0-15 hold K 0-15, lanes 16-31
        // hold K 16-31 of row M=ln (matches contiguous B layout).
        const bf16* arow = &U[cur][ln][kb * 32 + half * 16];
        v8bf alo = *(const v8bf*)(arow);
        v8bf ahi = *(const v8bf*)(arow + 8);
        v16bf a = __builtin_shufflevector(alo, ahi,
                   0,1,2,3,4,5,6,7,8,9,10,11,12,13,14,15);
        v16bf b1 = *(const v16bf*)(&PT[o1 + kb * 32 + half * 16]);
        acc0 = __builtin_amdgcn_wmma_f32_16x16x32_bf16(
                 false, a, false, Bres[kb], (short)0, acc0, false, false);
        acc1 = __builtin_amdgcn_wmma_f32_16x16x32_bf16(
                 false, a, false, b1, (short)0, acc1, false, false);
      }
      #pragma unroll
      for (int i = 0; i < 8; ++i) {
        const int m = i + 8 * half;
        tv0[i] = acc0[i] * et[m][col0];
        tv1[i] = acc1[i] * et[m][col1];
      }
    }

    // Row-sum reduction: xor-shuffle within each 16-lane half, then LDS add.
    #pragma unroll
    for (int i = 0; i < 8; ++i) {
      float s = tv0[i] + tv1[i];
      s += __shfl_xor(s, 1, 32);
      s += __shfl_xor(s, 2, 32);
      s += __shfl_xor(s, 4, 32);
      s += __shfl_xor(s, 8, 32);
      if (ln == 0) atomicAdd(&r[i + 8 * half], s);
    }
    __syncthreads();

    const int nxt = cur ^ 1;
    #pragma unroll
    for (int i = 0; i < 8; ++i) {
      const int m = i + 8 * half;
      const float inv = 1.f / r[m];
      U[nxt][m][col0] = (bf16)(tv0[i] * inv);
      U[nxt][m][col1] = (bf16)(tv1[i] * inv);
    }
    __syncthreads();

    if (tid < 16) {
      Mrow[tid] += __logf(r[tid]);
      sums[(size_t)t * B_ + bbase + tid] = Mrow[tid];  // logsumexp(alpha_t)
      r[tid] = 0.f;
    }
    __syncthreads();
    cur = nxt;
  }
}

// ---------------------------------------------------------------------------
// K5: log_probs[b] = sums[lengths[b]-1, b]
// ---------------------------------------------------------------------------
__global__ void finalize_kernel(const int* __restrict__ lengths,
                                const float* __restrict__ sums,
                                float* __restrict__ out) {
  const int b = threadIdx.x;
  if (b < B_) {
    const int len = lengths[b];
    out[b] = sums[(size_t)(len - 1) * B_ + b];
  }
}

// ---------------------------------------------------------------------------
extern "C" void kernel_launch(void* const* d_in, const int* in_sizes, int n_in,
                              void* d_out, int out_size, void* d_ws, size_t ws_size,
                              hipStream_t stream) {
  const int*   obs     = (const int*)d_in[0];    // (B,T,S) int
  const int*   lengths = (const int*)d_in[1];    // (B,) int
  const float* emis    = (const float*)d_in[2];  // (S,H,V) f32
  const float* tran    = (const float*)d_in[3];  // (H,H) f32
  const float* prior   = (const float*)d_in[4];  // (H,) f32
  float* out = (float*)d_out;

  char* ws = (char*)d_ws;
  float* lse  = (float*)(ws);                    //   8 KB @ 0
  float* sums = (float*)(ws + (16  << 10));      // 128 KB @ 16KB
  bf16*  PT   = (bf16*) (ws + (256 << 10));      // 512 KB @ 256KB
  float* eem  = (float*)(ws + (1   << 20));      //  64 MB @ 1MB

  emis_lse_kernel   <<<S_ * H_, 256, 0, stream>>>(emis, lse);
  tran_softmax_kernel<<<H_,     256, 0, stream>>>(tran, PT);
  emission_kernel   <<<B_ * T_, 256, 0, stream>>>(obs, emis, lse, eem);
  scan_kernel       <<<B_ / 16, 512, 0, stream>>>(PT, eem, prior, sums);
  finalize_kernel   <<<1,       64,  0, stream>>>(lengths, sums, out);
}
